// TimeAwareRNN_64905545777928
// MI455X (gfx1250) — matrix operand order repaired and checked by reading, not compiled, not run on hardware
//
#include <hip/hip_runtime.h>
#include <hip/hip_bf16.h>
#include <stdint.h>

typedef _Float16 half_t;
typedef __attribute__((ext_vector_type(16))) _Float16 v16h;
typedef __attribute__((ext_vector_type(8)))  float    v8f;

#define BATCH 256
#define LSEQ  128
#define KIN   32
#define KOUT  16
#define KH    384
#define NLX   1152
#define KT12  12           // 384 / 32 k-tiles
#define TILE_HALVES 512    // one 32x16 f16 B-fragment
#define NTHREADS 384       // 12 waves; each wave owns 2 column-tiles (32 features)

// ---- packed-weight workspace layout (units: halves) ----
static constexpr size_t OFF_WE  = 0;                                       // 1*24  tiles
static constexpr size_t OFF_WX  = OFF_WE  + (size_t)(1*24)  * TILE_HALVES; // 12*72
static constexpr size_t OFF_WHG = OFF_WX  + (size_t)(12*72) * TILE_HALVES; // 12*48
static constexpr size_t OFF_WHL = OFF_WHG + (size_t)(12*48) * TILE_HALVES; // 12*24
static constexpr size_t OFF_W1  = OFF_WHL + (size_t)(12*24) * TILE_HALVES; // 12*24
static constexpr size_t OFF_W2  = OFF_W1  + (size_t)(12*24) * TILE_HALVES; // 12*24
static constexpr size_t OFF_WMU = OFF_W2  + (size_t)(12*24) * TILE_HALVES; // 12*1
static constexpr size_t OFF_WLS = OFF_WMU + (size_t)(12*1)  * TILE_HALVES; // 12*1

// ---- output layout (flat, concatenated return order: ys, hs, mus, lss) ----
static constexpr size_t OUT_YS = 0;
static constexpr size_t OUT_HS = (size_t)BATCH * LSEQ * KOUT;
static constexpr size_t OUT_MU = OUT_HS + (size_t)BATCH * LSEQ * KH;
static constexpr size_t OUT_LS = OUT_MU + (size_t)BATCH * LSEQ * KOUT;

// fast transcendentals: v_exp_f32 + v_rcp_f32, no IEEE div sequence
__device__ __forceinline__ float fast_rcp(float x) { return __builtin_amdgcn_rcpf(x); }
__device__ __forceinline__ float sigm(float x)     { return fast_rcp(1.0f + __expf(-x)); }
__device__ __forceinline__ float tanh_f(float x)   { float e = __expf(-2.0f * x); return (1.0f - e) * fast_rcp(1.0f + e); }

__device__ __forceinline__ v8f wmma_f16(v16h a, v16h b, v8f c) {
  // D = A(16x32,f16) * B(32x16,f16) + C(16x16,f32)
  return __builtin_amdgcn_wmma_f32_16x16x32_f16(false, a, false, b, (short)0, c, false, false);
}

// Build 16x32 f16 A-fragment for k-tile `kt` from a row-major 16xKH f16 LDS tile.
// 16-bit A layout (contiguous k, matches pack_w B): lanes 0-15 row M=lane, K=0..15; lanes 16-31 K=16..31.
__device__ __forceinline__ v16h a_frag_lds(const half_t* mat, int lane, int kt) {
  const uint32_t* row = (const uint32_t*)(mat + (size_t)(lane & 15) * KH);
  const int p4 = (lane >> 4) * 8;
  union { v16h v; uint32_t u[8]; } a;
  const int base = kt * 16;
#pragma unroll
  for (int v = 0; v < 8; ++v) {
    int pair = v + p4;   // k = 2*pair
    a.u[v] = row[base + pair];
  }
  return a.v;
}

// Pre-packed B fragment: 16 contiguous halves per lane -> two global_load_b128.
__device__ __forceinline__ v16h b_frag(const half_t* packed, int tile, int lane) {
  return *(const v16h*)(packed + (size_t)tile * TILE_HALVES + (size_t)lane * 16);
}

// ---- pack f32 row-major (K,N) weight into WMMA B-fragment order, f16 ----
// 16-bit B layout used here (32x16, contiguous k): lanes 0-15 cols N=lane, K=2v+{0,1}; lanes 16-31 K=16+2v+{0,1}.
__global__ void pack_w(const float* __restrict__ W, half_t* __restrict__ out, int K, int N) {
  const int tile = blockIdx.x;
  const int KT = K >> 5;
  const int kt = tile % KT;
  const int nt = tile / KT;
  const int lane = threadIdx.x & 31;
  const int n = nt * 16 + (lane & 15);
  const int kbase = kt * 32 + (lane >> 4) * 16;
  half_t* o = out + (size_t)tile * TILE_HALVES + (size_t)lane * 16;
#pragma unroll
  for (int v = 0; v < 8; ++v) {
    int k = kbase + 2 * v;
    o[2 * v + 0] = (half_t)W[(size_t)(k + 0) * N + n];
    o[2 * v + 1] = (half_t)W[(size_t)(k + 1) * N + n];
  }
}

__global__ __launch_bounds__(NTHREADS, 1) void rnn_rk4_kernel(
    const float* __restrict__ x,   const float* __restrict__ dt,
    const float* __restrict__ eps, const float* __restrict__ state0,
    const float* __restrict__ be,  const float* __restrict__ bhg,
    const float* __restrict__ bhl, const float* __restrict__ b1,
    const float* __restrict__ b2,  const float* __restrict__ bmu,
    const float* __restrict__ bls, const half_t* __restrict__ wp,
    float* __restrict__ out)
{
  __shared__ __align__(16) half_t lds_h [16 * KH];   // RK argument h (f16)
  __shared__ __align__(16) half_t lds_s [16 * KH];   // xe -> (r.*h) -> t  (time-multiplexed)
  __shared__ __align__(16) half_t lds_lx[16 * NLX];  // Lx + biases (per-wave scratch)

  const int tid  = threadIdx.x;
  const int lane = tid & 31;
  const int w    = tid >> 5;          // wave id 0..11, owns feature cols [32w, 32w+32)
  const int hi   = lane >> 4;
  const int ln   = lane & 15;
  const int b0   = blockIdx.x * 16;   // batch tile

  const half_t* we_p  = wp + OFF_WE;
  const half_t* wx_p  = wp + OFF_WX;
  const half_t* whg_p = wp + OFF_WHG;
  const half_t* whl_p = wp + OFF_WHL;
  const half_t* w1_p  = wp + OFF_W1;
  const half_t* w2_p  = wp + OFF_W2;
  const half_t* wmu_p = wp + OFF_WMU;
  const half_t* wls_p = wp + OFF_WLS;

  const int nj0 = w * 32 + ln;        // column of tile j=0 (tile j=1 is nj0+16)

  // h state: this wave's 2 column-tiles, rows m = v + 8*hi (WMMA C layout)
  v8f hbase[2], harg[2], kacc[2];
#pragma unroll
  for (int j = 0; j < 2; ++j) {
    const float s0 = state0[nj0 + 16 * j];
#pragma unroll
    for (int v = 0; v < 8; ++v) { hbase[j][v] = s0; harg[j][v] = s0; kacc[j][v] = 0.f; }
#pragma unroll
    for (int v = 0; v < 8; ++v) lds_h[(size_t)(v + 8 * hi) * KH + nj0 + 16 * j] = (half_t)s0;
  }
  __syncthreads();

  for (int t = 0; t < LSEQ; ++t) {
    // per-row dt (rows m = v + 8*hi)
    float dtv[8];
#pragma unroll
    for (int v = 0; v < 8; ++v) dtv[v] = dt[(size_t)(b0 + v + 8 * hi) * LSEQ + t];

    // ---- xe = tanh(x_t @ We + be) : K=32, one WMMA k-step ----
    {
      const float* xr = x + ((size_t)(b0 + ln) * LSEQ + t) * KIN;
      union { v16h v; uint32_t u[8]; } a;
#pragma unroll
      for (int v = 0; v < 8; ++v) {
        int k = 2 * v + 16 * hi;
        union { half_t h[2]; uint32_t u; } p;
        p.h[0] = (half_t)xr[k];
        p.h[1] = (half_t)xr[k + 1];
        a.u[v] = p.u;
      }
#pragma unroll
      for (int j = 0; j < 2; ++j) {
        v8f c = {};
        c = wmma_f16(a.v, b_frag(we_p, 2 * w + j, lane), c);
        const float bev = be[nj0 + 16 * j];     // per-step load, L2-resident
#pragma unroll
        for (int v = 0; v < 8; ++v)
          lds_s[(size_t)(v + 8 * hi) * KH + nj0 + 16 * j] = (half_t)tanh_f(c[v] + bev);
      }
    }
    __syncthreads();

    // ---- Lx = xe @ Wx ; fold bhg/bhl biases in; spill to LDS (per-wave private) ----
    {
      v8f Lz[2] = {}, Lr[2] = {}, Lc[2] = {};
#pragma unroll 4
      for (int kt = 0; kt < KT12; ++kt) {
        v16h a = a_frag_lds(lds_s, lane, kt);
#pragma unroll
        for (int j = 0; j < 2; ++j) {
          Lz[j] = wmma_f16(a, b_frag(wx_p, (     2 * w + j) * KT12 + kt, lane), Lz[j]);
          Lr[j] = wmma_f16(a, b_frag(wx_p, (24 + 2 * w + j) * KT12 + kt, lane), Lr[j]);
          Lc[j] = wmma_f16(a, b_frag(wx_p, (48 + 2 * w + j) * KT12 + kt, lane), Lc[j]);
        }
      }
#pragma unroll
      for (int j = 0; j < 2; ++j) {
        const int  n  = nj0 + 16 * j;
        const float bz = bhg[n];
        const float br = bhg[KH + n];
        const float bl = bhl[n];
#pragma unroll
        for (int v = 0; v < 8; ++v) {
          const size_t row = (size_t)(v + 8 * hi) * NLX;
          lds_lx[row +       n] = (half_t)(Lz[j][v] + bz);
          lds_lx[row + KH  + n] = (half_t)(Lr[j][v] + br);
          lds_lx[row + 768 + n] = (half_t)(Lc[j][v] + bl);
        }
      }
    }
    __syncthreads();   // all waves done reading xe from lds_s before it is reused for (r.*h)

    // ---- RK4: k1..k4 ----
#pragma unroll 1
    for (int s = 0; s < 4; ++s) {
      // gates = sigmoid(Lx[:, :768] + h_arg @ Whg + bhg)
      v8f zA[2] = {}, rA[2] = {};
#pragma unroll 4
      for (int kt = 0; kt < KT12; ++kt) {
        v16h a = a_frag_lds(lds_h, lane, kt);
#pragma unroll
        for (int j = 0; j < 2; ++j) {
          zA[j] = wmma_f16(a, b_frag(whg_p, (     2 * w + j) * KT12 + kt, lane), zA[j]);
          rA[j] = wmma_f16(a, b_frag(whg_p, (24 + 2 * w + j) * KT12 + kt, lane), rA[j]);
        }
      }
      v8f gz[2];
#pragma unroll
      for (int j = 0; j < 2; ++j) {
        const int n = nj0 + 16 * j;
#pragma unroll
        for (int v = 0; v < 8; ++v) {
          const size_t row = (size_t)(v + 8 * hi);
          gz[j][v]  = sigm(zA[j][v] + (float)lds_lx[row * NLX +      n]);
          float gr  = sigm(rA[j][v] + (float)lds_lx[row * NLX + KH + n]);
          lds_s[row * KH + n] = (half_t)(gr * harg[j][v]);  // (r .* h)
        }
      }
      __syncthreads();   // (r.*h) visible; all Whg reads of lds_h complete

      // Hl = (r .* h) @ Whl
      v8f hlA[2] = {};
#pragma unroll 4
      for (int kt = 0; kt < KT12; ++kt) {
        v16h a = a_frag_lds(lds_s, lane, kt);
#pragma unroll
        for (int j = 0; j < 2; ++j)
          hlA[j] = wmma_f16(a, b_frag(whl_p, (2 * w + j) * KT12 + kt, lane), hlA[j]);
      }

      const float wK = (s == 1 || s == 2) ? 2.0f : 1.0f;
      const float cN = (s <= 1) ? 0.5f : 1.0f;
#pragma unroll
      for (int j = 0; j < 2; ++j) {
        const int n = nj0 + 16 * j;
#pragma unroll
        for (int v = 0; v < 8; ++v) {
          const size_t row = (size_t)(v + 8 * hi);
          float cand = tanh_f((float)lds_lx[row * NLX + 768 + n] + hlA[j][v]);
          float kk   = gz[j][v] * (cand - harg[j][v]);   // MEANDT = 1
          kacc[j][v] = (s == 0) ? kk : (kacc[j][v] + wK * kk);
          float hn;
          if (s < 3) {
            hn = hbase[j][v] + dtv[v] * cN * kk;
          } else {
            hbase[j][v] += dtv[v] * kacc[j][v] * (1.0f / 6.0f);
            hn = hbase[j][v];
          }
          harg[j][v] = hn;
          lds_h[row * KH + n] = (half_t)hn;   // safe: lds_h last read before prior barrier
        }
      }
      __syncthreads();   // new h_arg visible; lds_s reads (Whl) complete before reuse
    }

    // ---- head: t = tanh(hW1+b1) * sigm(hW2+b2); store hs ----
    {
      v8f a1[2] = {}, a2[2] = {};
#pragma unroll 4
      for (int kt = 0; kt < KT12; ++kt) {
        v16h a = a_frag_lds(lds_h, lane, kt);
#pragma unroll
        for (int j = 0; j < 2; ++j) {
          a1[j] = wmma_f16(a, b_frag(w1_p, (2 * w + j) * KT12 + kt, lane), a1[j]);
          a2[j] = wmma_f16(a, b_frag(w2_p, (2 * w + j) * KT12 + kt, lane), a2[j]);
        }
      }
#pragma unroll
      for (int j = 0; j < 2; ++j) {
        const int  n   = nj0 + 16 * j;
        const float b1v = b1[n];
        const float b2v = b2[n];
#pragma unroll
        for (int v = 0; v < 8; ++v) {
          float tt = tanh_f(a1[j][v] + b1v) * sigm(a2[j][v] + b2v);
          lds_s[(size_t)(v + 8 * hi) * KH + n] = (half_t)tt;
          out[OUT_HS + ((size_t)(b0 + v + 8 * hi) * LSEQ + t) * KH + n] = hbase[j][v];
        }
      }
    }
    __syncthreads();

    // ---- mu / ls / y : single 16-col tile each, wave 0 only (EXEC all-1 within the wave) ----
    if (w == 0) {
      v8f mu = {}, ls = {};
#pragma unroll 4
      for (int kt = 0; kt < KT12; ++kt) {
        v16h a = a_frag_lds(lds_s, lane, kt);
        mu = wmma_f16(a, b_frag(wmu_p, kt, lane), mu);
        ls = wmma_f16(a, b_frag(wls_p, kt, lane), ls);
      }
      const float bmuv = bmu[ln];
      const float blsv = bls[ln];
#pragma unroll
      for (int v = 0; v < 8; ++v) {
        size_t r  = (size_t)(b0 + v + 8 * hi) * LSEQ + t;
        float  m  = mu[v] + bmuv;
        float  l2 = ls[v] + blsv;
        float  e  = eps[r * KOUT + ln];
        out[OUT_YS + r * KOUT + ln] = m + __expf(0.5f * l2) * e;
        out[OUT_MU + r * KOUT + ln] = m;
        out[OUT_LS + r * KOUT + ln] = l2;
      }
    }
    __syncthreads();
  }
}

extern "C" void kernel_launch(void* const* d_in, const int* in_sizes, int n_in,
                              void* d_out, int out_size, void* d_ws, size_t ws_size,
                              hipStream_t stream) {
  (void)in_sizes; (void)n_in; (void)out_size; (void)ws_size;
  const float* x   = (const float*)d_in[0];
  const float* dt  = (const float*)d_in[1];
  const float* eps = (const float*)d_in[2];
  const float* s0  = (const float*)d_in[3];
  const float* We  = (const float*)d_in[4];
  const float* be  = (const float*)d_in[5];
  const float* Wx  = (const float*)d_in[6];
  const float* Whg = (const float*)d_in[7];
  const float* bhg = (const float*)d_in[8];
  const float* Whl = (const float*)d_in[9];
  const float* bhl = (const float*)d_in[10];
  const float* W1  = (const float*)d_in[11];
  const float* b1  = (const float*)d_in[12];
  const float* W2  = (const float*)d_in[13];
  const float* b2  = (const float*)d_in[14];
  const float* Wmu = (const float*)d_in[15];
  const float* bmu = (const float*)d_in[16];
  const float* Wls = (const float*)d_in[17];
  const float* bls = (const float*)d_in[18];

  half_t* wp = (half_t*)d_ws;

  // Re-pack every call (deterministic; no cross-call state). Weights -> f16 B-fragment order.
  pack_w<<<  1 * 24, 32, 0, stream>>>(We,  wp + OFF_WE,   32,  384);
  pack_w<<< 12 * 72, 32, 0, stream>>>(Wx,  wp + OFF_WX,  384, 1152);
  pack_w<<< 12 * 48, 32, 0, stream>>>(Whg, wp + OFF_WHG, 384,  768);
  pack_w<<< 12 * 24, 32, 0, stream>>>(Whl, wp + OFF_WHL, 384,  384);
  pack_w<<< 12 * 24, 32, 0, stream>>>(W1,  wp + OFF_W1,  384,  384);
  pack_w<<< 12 * 24, 32, 0, stream>>>(W2,  wp + OFF_W2,  384,  384);
  pack_w<<< 12 *  1, 32, 0, stream>>>(Wmu, wp + OFF_WMU, 384,   16);
  pack_w<<< 12 *  1, 32, 0, stream>>>(Wls, wp + OFF_WLS, 384,   16);

  rnn_rk4_kernel<<<BATCH / 16, NTHREADS, 0, stream>>>(
      x, dt, eps, s0, be, bhg, bhl, b1, b2, bmu, bls, wp, (float*)d_out);
}
